// TorchMD_Norm_Hami_43499428774268
// MI455X (gfx1250) — matrix operand order repaired and checked by reading, not compiled, not run on hardware
//
#include <hip/hip_runtime.h>
#include <hip/hip_bf16.h>
#include <math.h>

#define H 128

typedef __attribute__((ext_vector_type(16))) _Float16 v16h;
typedef __attribute__((ext_vector_type(8)))  _Float16 v8h;
typedef __attribute__((ext_vector_type(8)))  float    v8f;
typedef __attribute__((ext_vector_type(4)))  unsigned int u32x4;
typedef __attribute__((ext_vector_type(8)))  int      i32x8;
typedef __attribute__((ext_vector_type(4)))  int      i32x4;

#if defined(__has_builtin)
#if __has_builtin(__builtin_amdgcn_tensor_load_to_lds)
#define HAVE_TDM 1
#endif
#endif
#if defined(HAVE_TDM)
#if __has_include(<hip/amd_detail/amd_gfx1250_TDM.h>)
#define TDM_ARGS6 1
#endif
#endif

static __device__ __forceinline__ float silu_f(float x) { return x / (1.f + __expf(-x)); }

// ---- packed-weight layout (halves). B-fragment lane order, see k_pack_w --
// 8 x [128x128]: Wq Wk Wv Wdk Wdv Wf Wsrc Wtrg, then Ws [256x128],
// Wvec [384x128], Wo [384x128]
#define PK_Q    0
#define PK_K    16384
#define PK_V    32768
#define PK_DK   49152
#define PK_DV   65536
#define PK_F    81920
#define PK_SRC  98304
#define PK_TRG  114688
#define PK_S    131072
#define PK_VEC  163840
#define PK_O    212992
#define PK_TOT  262144

// ---- WMMA fragment helpers (V_WMMA_F32_16X16X32_F16, wave32) -------------
// A 16x32 f16: lanes 0-15 -> row m, K=[k0..k0+7]+[k0+16..k0+23];
// lanes 16-31 -> row m, K=[k0+8..k0+15]+[k0+24..k0+31].
__device__ __forceinline__ v16h load_afrag_f32(const float* base, const int* rowoff,
                                               int rowadd, int k0, int lane) {
  const int m = lane & 15;
  const int hf = (lane >> 4) * 8;
  const float* r = base + rowoff[m] + rowadd + k0 + hf;
  v16h f;
#pragma unroll
  for (int i = 0; i < 8; ++i) {
    f[i]     = (_Float16)r[i];
    f[i + 8] = (_Float16)r[16 + i];
  }
  return f;
}

// contiguous f32 rows (LDS staging buffers)
__device__ __forceinline__ v16h load_afrag_lds(const float* base, int ld, int k0, int lane) {
  const int m = lane & 15;
  const int hf = (lane >> 4) * 8;
  const float* r = base + (size_t)m * ld + k0 + hf;
  v16h f;
#pragma unroll
  for (int i = 0; i < 8; ++i) {
    f[i]     = (_Float16)r[i];
    f[i + 8] = (_Float16)r[16 + i];
  }
  return f;
}

// f16 gathered rows: two 16B vector loads, no conversions
__device__ __forceinline__ v16h load_afrag_h(const _Float16* base, const int* rowoff,
                                             int rowadd, int k0, int lane) {
  const int m = lane & 15;
  const int hf = (lane >> 4) * 8;
  const _Float16* r = base + rowoff[m] + rowadd + k0 + hf;
  v8h lo = *(const v8h*)(r);
  v8h hi = *(const v8h*)(r + 16);
  v16h f;
#pragma unroll
  for (int i = 0; i < 8; ++i) { f[i] = lo[i]; f[8 + i] = hi[i]; }
  return f;
}

// packed B frag: one 32-byte per-lane load (coalesced 1KB per wave)
__device__ __forceinline__ v16h load_bfrag_pk(const _Float16* Wpk, int n0, int kt, int lane) {
  return ((const v16h*)Wpk)[((((n0 >> 4) << 2) + kt) << 5) + lane];
}

__device__ __forceinline__ v8f wmma4(const v16h a[4], const _Float16* Wpk, int n0, int lane) {
  v8f c = {};
#pragma unroll
  for (int kt = 0; kt < 4; ++kt) {
    v16h b = load_bfrag_pk(Wpk, n0, kt, lane);
    c = __builtin_amdgcn_wmma_f32_16x16x32_f16(false, a[kt], false, b,
                                               (short)0, c, false, false);
  }
  return c;
}

// D layout: reg r, lane l -> row = r + 8*(l>>4), col = (l&15)
__device__ __forceinline__ void store_tile(float* dst, int ldc, int col0, v8f c,
                                           int lane, const float* bias) {
  const int n = lane & 15;
  const int rb = (lane >> 4) * 8;
  float b = bias ? bias[n] : 0.f;
#pragma unroll
  for (int r = 0; r < 8; ++r)
    dst[(size_t)(r + rb) * ldc + col0 + n] = c[r] + b;
}

__device__ __forceinline__ void store_tile_g(float* dst, int ldc, int col0, v8f c,
                                             int lane, const float* bias, int rows_valid) {
  const int n = lane & 15;
  const int rb = (lane >> 4) * 8;
  float b = bias ? bias[n] : 0.f;
#pragma unroll
  for (int r = 0; r < 8; ++r) {
    int m = r + rb;
    if (m < rows_valid) dst[(size_t)m * ldc + col0 + n] = c[r] + b;
  }
}

// ---- Kernel 0: pack weights to f16 fragment order ------------------------
__global__ __launch_bounds__(256) void k_pack_w(
    const float* Wq, const float* Wk, const float* Wv, const float* Wdk,
    const float* Wdv, const float* Wf, const float* Wsrc, const float* Wtrg,
    const float* Wsm, const float* Wvec, const float* Wo, _Float16* pk) {
  int gid = blockIdx.x * 256 + threadIdx.x;   // one 16-half group per thread
  if (gid >= (PK_TOT >> 4)) return;
  int off = gid << 4;
  const float* W; int rel;
  if (off < PK_S) {
    int m = off >> 14; rel = off & 16383;
    const float* tab[8] = {Wq, Wk, Wv, Wdk, Wdv, Wf, Wsrc, Wtrg};
    W = tab[m];
  } else if (off < PK_VEC) { rel = off - PK_S;   W = Wsm;  }
  else if (off < PK_O)     { rel = off - PK_VEC; W = Wvec; }
  else                     { rel = off - PK_O;   W = Wo;   }
  int g = rel >> 4;
  int lane = g & 31, tk = g >> 5, kt = tk & 3, tile = tk >> 2;
  const float* src = W + (size_t)(tile * 16 + (lane & 15)) * H + kt * 32 + (lane >> 4) * 8;
  _Float16* dst = pk + off;
#pragma unroll
  for (int i = 0; i < 16; ++i) dst[i] = (_Float16)src[i < 8 ? i : 8 + i];
}

// ---- Kernel 1: node layernorm + vector max-min norm ----------------------
__global__ __launch_bounds__(128) void k_node_norm(
    const float* __restrict__ x, const float* __restrict__ vec,
    const float* __restrict__ g, const float* __restrict__ b,
    _Float16* __restrict__ xn_h, float* __restrict__ vecn,
    _Float16* __restrict__ vecn_h, int N) {
  __shared__ float red[128];
  const int n = blockIdx.x, t = threadIdx.x;
  if (n >= N) return;

  float xv = x[(size_t)n * H + t];
  red[t] = xv; __syncthreads();
  for (int s = 64; s > 0; s >>= 1) { if (t < s) red[t] += red[t + s]; __syncthreads(); }
  float mean = red[0] * (1.f / H); __syncthreads();
  float dv = xv - mean;
  red[t] = dv * dv; __syncthreads();
  for (int s = 64; s > 0; s >>= 1) { if (t < s) red[t] += red[t + s]; __syncthreads(); }
  float var = red[0] * (1.f / H); __syncthreads();
  xn_h[(size_t)n * H + t] = (_Float16)(dv * rsqrtf(var + 1e-5f) * g[t] + b[t]);

  float v0 = vec[(size_t)n * 384 + t];
  float v1 = vec[(size_t)n * 384 + 128 + t];
  float v2 = vec[(size_t)n * 384 + 256 + t];
  float dist = sqrtf(v0 * v0 + v1 * v1 + v2 * v2);
  dist = fmaxf(dist, 1e-8f);
  red[t] = dist; __syncthreads();
  for (int s = 64; s > 0; s >>= 1) { if (t < s) red[t] = fmaxf(red[t], red[t + s]); __syncthreads(); }
  float mx = red[0]; __syncthreads();
  red[t] = dist; __syncthreads();
  for (int s = 64; s > 0; s >>= 1) { if (t < s) red[t] = fminf(red[t], red[t + s]); __syncthreads(); }
  float mn = red[0]; __syncthreads();
  float delta = (mx - mn == 0.f) ? 1.f : (mx - mn);
  float scale = fmaxf((dist - mn) / delta, 0.f) / dist;
  float o0 = v0 * scale, o1 = v1 * scale, o2 = v2 * scale;
  vecn[(size_t)n * 384 + t]       = o0;
  vecn[(size_t)n * 384 + 128 + t] = o1;
  vecn[(size_t)n * 384 + 256 + t] = o2;
  vecn_h[(size_t)n * 384 + t]       = (_Float16)o0;
  vecn_h[(size_t)n * 384 + 128 + t] = (_Float16)o1;
  vecn_h[(size_t)n * 384 + 256 + t] = (_Float16)o2;
}

// ---- Kernel 2: qkv GEMM + Wvec projection (vec_dot, vec3) ----------------
__global__ __launch_bounds__(128) void k_node_qkv(
    const _Float16* __restrict__ xn_h, const _Float16* __restrict__ vecn_h,
    const _Float16* __restrict__ pk,
    const float* bq, const float* bk, const float* bv,
    float* __restrict__ qkv, float* __restrict__ vec_dot,
    float* __restrict__ vec3, int N) {
  __shared__ float sbuf[16 * 384];
  __shared__ float vdacc[16 * 128];
  __shared__ int roff[16], roffv[16];
  const int t = threadIdx.x, wave = t >> 5, lane = t & 31;
  const int tile0 = blockIdx.x * 16;
  const int rows_valid = min(16, N - tile0);
  if (t < 16) {
    int nn = min(tile0 + t, N - 1);
    roff[t] = nn * H;
    roffv[t] = nn * 384;
  }
  for (int i = t; i < 2048; i += 128) vdacc[i] = 0.f;
  __syncthreads();

  v16h afr[4];
#pragma unroll
  for (int kt = 0; kt < 4; ++kt) afr[kt] = load_afrag_h(xn_h, roff, 0, kt * 32, lane);
  for (int j = wave; j < 24; j += 4) {
    const _Float16* Wpk = pk + ((j < 8) ? PK_Q : (j < 16 ? PK_K : PK_V));
    const float* bb = (j < 8) ? bq : (j < 16 ? bk : bv);
    int n0 = (j & 7) * 16;
    v8f c = wmma4(afr, Wpk, n0, lane);
    store_tile_g(qkv + (size_t)tile0 * 384, 384, j * 16, c, lane, bb + n0, rows_valid);
  }

  for (int dim = 0; dim < 3; ++dim) {
#pragma unroll
    for (int kt = 0; kt < 4; ++kt)
      afr[kt] = load_afrag_h(vecn_h, roffv, dim * 128, kt * 32, lane);
    for (int j = wave; j < 24; j += 4) {
      v8f c = wmma4(afr, pk + PK_VEC, j * 16, lane);
      store_tile(sbuf, 384, j * 16, c, lane, nullptr);
    }
    __syncthreads();
    for (int idx = t; idx < 2048; idx += 128) {
      int e = idx >> 7, cc = idx & 127;
      vdacc[idx] += sbuf[e * 384 + cc] * sbuf[e * 384 + 128 + cc];
      int node = tile0 + e;
      if (node < N) vec3[(size_t)node * 384 + dim * 128 + cc] = sbuf[e * 384 + 256 + cc];
    }
    __syncthreads();
  }
  for (int idx = t; idx < 2048; idx += 128) {
    int e = idx >> 7, cc = idx & 127;
    int node = tile0 + e;
    if (node < N) vec_dot[(size_t)node * H + cc] = vdacc[idx];
  }
}

// ---- Kernel 3: zero accumulators ----------------------------------------
__global__ __launch_bounds__(256) void k_zero(float* p, size_t n) {
  size_t i = (size_t)blockIdx.x * 256 + threadIdx.x;
  size_t stride = (size_t)gridDim.x * 256;
  for (; i < n; i += stride) p[i] = 0.f;
}

// ---- Kernel 4: fused edge kernel -----------------------------------------
__global__ __launch_bounds__(128) void k_edge(
    const float* __restrict__ qkv, const float* __restrict__ vecn,
    const _Float16* __restrict__ vecn_h,
    const float* __restrict__ f_ij, const int* __restrict__ eidx,
    const float* __restrict__ r_ij, const float* __restrict__ d_ij,
    const _Float16* __restrict__ pk,
    const float* bdk, const float* bdv, const float* bf, const float* bs,
    float* __restrict__ x_agg, float* __restrict__ vec_agg, float* __restrict__ df_out,
    int E) {
  // regions: 0:dk/wt 2048:dv/ws 4096:fpre 6144:vj/accA 8192:fij/s/accPt 10240:s/accPs
  __shared__ float smem[12288];
  __shared__ int s_src[16], s_dst[16], s_esrc[16], s_edst[16];
  __shared__ float s_r[16], s_dij[48], s_attn[128];
  const int t = threadIdx.x, wave = t >> 5, lane = t & 31;
  const long e0 = (long)blockIdx.x * 16;
  const int nvalid = (int)min((long)16, (long)E - e0);
  float* s_fij = smem + 8192;   // TDM landing zone (freed before phase 3)

  if (t < 16) {
    long e = e0 + t; if (e >= E) e = E - 1;
    int sj = eidx[e], di = eidx[(size_t)E + e];
    s_src[t] = sj; s_dst[t] = di;
    s_esrc[t] = sj * 384; s_edst[t] = di * 384;
    s_r[t] = r_ij[e];
  }
  if (t < 48) {
    long e = e0 + (t / 3); if (e >= E) e = E - 1;
    s_dij[t] = d_ij[e * 3 + (t % 3)];
  }
  if (t == 0 && e0 + 16 < E) __builtin_prefetch(f_ij + (e0 + 16) * H, 0, 1);

  // ---- TDM: async DMA f_ij tile [16 x 128] f32 -> LDS --------------------
#if defined(HAVE_TDM)
  if (wave == 0) {
    unsigned long long gaddr = (unsigned long long)(uintptr_t)f_ij
                             + (unsigned long long)e0 * (H * 4ull);
    unsigned int lds = (unsigned int)(uintptr_t)s_fij;
    u32x4 g0;
    g0[0] = 1u;                                      // count=1, user mode
    g0[1] = lds;                                     // lds_addr
    g0[2] = (unsigned int)(gaddr & 0xffffffffu);     // global_addr lo
    g0[3] = (unsigned int)((gaddr >> 32) & 0x1ffffffu) | (2u << 30); // hi|type=2
    i32x8 g1;
    g1[0] = (int)(2u << 16);                         // data_size = 4B
    g1[1] = (int)(128u << 16);                       // tensor_dim0 = 128 (lo16)
    g1[2] = (int)(((unsigned)E & 0xffffu) << 16);    // dim0 hi=0 | tensor_dim1 lo16 = E
    g1[3] = (int)((((unsigned)E >> 16) & 0xffffu) | (128u << 16)); // dim1 hi | tile_dim0=128
    g1[4] = 16;                                      // tile_dim1 = 16
    g1[5] = 128;                                     // tensor_dim0_stride = 128
    g1[6] = 0;
    g1[7] = 0;
    i32x4 z4 = {0, 0, 0, 0};
#if defined(TDM_ARGS6)
    i32x8 z8 = {0, 0, 0, 0, 0, 0, 0, 0};
    __builtin_amdgcn_tensor_load_to_lds(g0, g1, z4, z4, z8, 0);
#else
    __builtin_amdgcn_tensor_load_to_lds(g0, g1, z4, z4, 0);
#endif
    __builtin_amdgcn_s_wait_tensorcnt(0);
  }
#else
  __syncthreads();
  for (int idx = t; idx < 2048; idx += 128) {
    int e = idx >> 7; long ee = e0 + e; if (ee >= E) ee = E - 1;
    s_fij[idx] = f_ij[(size_t)ee * H + (idx & 127)];
  }
#endif
  __syncthreads();

  // Phase 1: dk = silu(f@Wdk^T+b), dv, fpre = silu(f@Wf^T+b)
  v16h afr[4];
#pragma unroll
  for (int kt = 0; kt < 4; ++kt) afr[kt] = load_afrag_lds(s_fij, H, kt * 32, lane);
  for (int j = wave; j < 24; j += 4) {
    int mtx = j >> 3, n0 = (j & 7) * 16;
    const _Float16* Wpk = pk + (mtx == 0 ? PK_DK : (mtx == 1 ? PK_DV : PK_F));
    const float* bb = (mtx == 0) ? bdk : (mtx == 1 ? bdv : bf);
    v8f c = wmma4(afr, Wpk, n0, lane);
    store_tile(smem + mtx * 2048, H, n0, c, lane, bb + n0);
  }
  __syncthreads();
  for (int i = t; i < 6144; i += 128) smem[i] = silu_f(smem[i]);
  __syncthreads();

  // Phase 2: attention + vj, scatter x_agg
  {
    int e = t >> 3, h = t & 7;
    const float* qr = qkv + (size_t)s_edst[e] + h * 16;
    const float* kr = qkv + (size_t)s_esrc[e] + 128 + h * 16;
    const float* dkr = smem + e * H + h * 16;
    float a = 0.f;
#pragma unroll
    for (int d = 0; d < 16; ++d) a += qr[d] * kr[d] * dkr[d];
    float r = s_r[e];
    float cut = 0.5f * (__cosf(0.62831853f * r) + 1.f) * (r < 5.f ? 1.f : 0.f);
    s_attn[t] = silu_f(a) * cut;
  }
  __syncthreads();
  for (int idx = t; idx < 2048; idx += 128) {
    int e = idx >> 7, cc = idx & 127;
    float val = qkv[(size_t)s_esrc[e] + 256 + cc] * smem[2048 + idx] * s_attn[e * 8 + (cc >> 4)];
    smem[6144 + idx] = val;
    if (e < nvalid) atomicAdd(&x_agg[(size_t)s_dst[e] * H + cc], val);
  }
  __syncthreads();

  // Phase 3: s = silu(vj @ Ws^T + bs)  [16 x 256]  (overwrites s_fij zone)
#pragma unroll
  for (int kt = 0; kt < 4; ++kt) afr[kt] = load_afrag_lds(smem + 6144, H, kt * 32, lane);
  for (int j = wave; j < 16; j += 4) {
    v8f c = wmma4(afr, pk + PK_S, j * 16, lane);
    store_tile(smem + 8192, 256, j * 16, c, lane, bs + j * 16);
  }
  __syncthreads();
  for (int i = t; i < 4096; i += 128) smem[8192 + i] = silu_f(smem[8192 + i]);
  __syncthreads();

  // Phase 4: vec_msg scatter to vec_agg
  for (int idx = t; idx < 6144; idx += 128) {
    int e = idx / 384, rem = idx - e * 384, dim = rem >> 7, cc = rem & 127;
    if (e < nvalid) {
      float val = vecn[(size_t)s_esrc[e] + dim * 128 + cc] * smem[8192 + e * 256 + cc]
                + smem[8192 + e * 256 + 128 + cc] * s_dij[e * 3 + dim];
      atomicAdd(&vec_agg[(size_t)s_edst[e] + dim * 128 + cc], val);
    }
  }
  __syncthreads();

  // Phase 5: edge update. w_dot = sum_d wt*ws - (wt.d)(ws.d)  (d is unit)
  for (int i = t; i < 6144; i += 128) smem[6144 + i] = 0.f;
  __syncthreads();
  for (int dim = 0; dim < 3; ++dim) {
    if (wave < 2) {
#pragma unroll
      for (int kt = 0; kt < 4; ++kt)
        afr[kt] = load_afrag_h(vecn_h, s_edst, dim * 128, kt * 32, lane);
      for (int j = wave; j < 8; j += 2) {
        v8f c = wmma4(afr, pk + PK_TRG, j * 16, lane);
        store_tile(smem, H, j * 16, c, lane, nullptr);
      }
    } else {
#pragma unroll
      for (int kt = 0; kt < 4; ++kt)
        afr[kt] = load_afrag_h(vecn_h, s_esrc, dim * 128, kt * 32, lane);
      for (int j = wave - 2; j < 8; j += 2) {
        v8f c = wmma4(afr, pk + PK_SRC, j * 16, lane);
        store_tile(smem + 2048, H, j * 16, c, lane, nullptr);
      }
    }
    __syncthreads();
    for (int idx = t; idx < 2048; idx += 128) {
      int e = idx >> 7;
      float wt = smem[idx], ws2 = smem[2048 + idx], dd = s_dij[e * 3 + dim];
      smem[6144 + idx]  += wt * ws2;
      smem[8192 + idx]  += wt * dd;
      smem[10240 + idx] += ws2 * dd;
    }
    __syncthreads();
  }
  for (int idx = t; idx < 2048; idx += 128) {
    int e = idx >> 7, cc = idx & 127;
    if (e < nvalid) {
      float wd = smem[6144 + idx] - smem[8192 + idx] * smem[10240 + idx];
      df_out[(size_t)(e0 + e) * H + cc] = smem[4096 + idx] * wd;
    }
  }
}

// ---- Kernel 5: output GEMM (Wo) + final elementwise ----------------------
__global__ __launch_bounds__(128) void k_node_out(
    const float* __restrict__ x_agg, const float* __restrict__ vec_agg,
    const float* __restrict__ vec_dot, const float* __restrict__ vec3,
    const _Float16* __restrict__ pk, const float* bo,
    float* __restrict__ dx, float* __restrict__ dvec, int N) {
  __shared__ float obuf[16 * 384];
  __shared__ int roff[16];
  const int t = threadIdx.x, wave = t >> 5, lane = t & 31;
  const int tile0 = blockIdx.x * 16;
  if (t < 16) roff[t] = min(tile0 + t, N - 1) * H;
  __syncthreads();

  v16h afr[4];
#pragma unroll
  for (int kt = 0; kt < 4; ++kt) afr[kt] = load_afrag_f32(x_agg, roff, 0, kt * 32, lane);
  for (int j = wave; j < 24; j += 4) {
    v8f c = wmma4(afr, pk + PK_O, j * 16, lane);
    store_tile(obuf, 384, j * 16, c, lane, bo + j * 16);
  }
  __syncthreads();

  for (int idx = t; idx < 2048; idx += 128) {
    int e = idx >> 7, cc = idx & 127;
    int node = tile0 + e;
    if (node < N)
      dx[(size_t)node * H + cc] =
          vec_dot[(size_t)node * H + cc] * obuf[e * 384 + 128 + cc] + obuf[e * 384 + 256 + cc];
  }
  for (int idx = t; idx < 6144; idx += 128) {
    int e = idx / 384, rem = idx - e * 384, cc = rem & 127;
    int node = tile0 + e;
    if (node < N)
      dvec[(size_t)node * 384 + rem] =
          vec3[(size_t)node * 384 + rem] * obuf[e * 384 + cc] + vec_agg[(size_t)node * 384 + rem];
  }
}

// ---- launch --------------------------------------------------------------
extern "C" void kernel_launch(void* const* d_in, const int* in_sizes, int n_in,
                              void* d_out, int out_size, void* d_ws, size_t ws_size,
                              hipStream_t stream) {
  const float* x     = (const float*)d_in[0];
  const float* vec   = (const float*)d_in[1];
  const int*   eidx  = (const int*)d_in[2];
  const float* r_ij  = (const float*)d_in[3];
  const float* f_ij  = (const float*)d_in[4];
  const float* d_ij  = (const float*)d_in[5];
  const float* ln_g  = (const float*)d_in[6];
  const float* ln_b  = (const float*)d_in[7];
  const float* Wq = (const float*)d_in[8];   const float* bq = (const float*)d_in[9];
  const float* Wk = (const float*)d_in[10];  const float* bk = (const float*)d_in[11];
  const float* Wv = (const float*)d_in[12];  const float* bv = (const float*)d_in[13];
  const float* Wsm = (const float*)d_in[14]; const float* bs = (const float*)d_in[15];
  const float* Wo = (const float*)d_in[16];  const float* bo = (const float*)d_in[17];
  const float* Wvec = (const float*)d_in[18];
  const float* Wdk = (const float*)d_in[19]; const float* bdk = (const float*)d_in[20];
  const float* Wdv = (const float*)d_in[21]; const float* bdv = (const float*)d_in[22];
  const float* Wf = (const float*)d_in[23];  const float* bf = (const float*)d_in[24];
  const float* Wsrc = (const float*)d_in[25];
  const float* Wtrg = (const float*)d_in[26];

  const int N = in_sizes[0] / H;
  const int E = in_sizes[2] / 2;

  float* ws = (float*)d_ws;
  size_t off = 0;
  float* vecn    = ws + off; off += (size_t)N * 3 * H;
  float* qkv     = ws + off; off += (size_t)N * 3 * H;
  float* vdot    = ws + off; off += (size_t)N * H;
  float* vec3    = ws + off; off += (size_t)N * 3 * H;
  float* x_agg   = ws + off; off += (size_t)N * H;       // x_agg + vec_agg contiguous
  float* vec_agg = ws + off; off += (size_t)N * 3 * H;
  _Float16* hbase  = (_Float16*)(ws + off);
  _Float16* xn_h   = hbase;                       // N*128 halves
  _Float16* vecn_h = hbase + (size_t)N * H;       // N*384 halves
  _Float16* pk     = vecn_h + (size_t)N * 3 * H;  // PK_TOT halves

  float* dx   = (float*)d_out;
  float* dvec = dx + (size_t)N * H;
  float* dfij = dvec + (size_t)N * 3 * H;

  const int ntiles = (N + 15) / 16;
  const int etiles = (E + 15) / 16;

  k_pack_w<<<(PK_TOT / 16 + 255) / 256, 256, 0, stream>>>(
      Wq, Wk, Wv, Wdk, Wdv, Wf, Wsrc, Wtrg, Wsm, Wvec, Wo, pk);
  k_node_norm<<<N, 128, 0, stream>>>(x, vec, ln_g, ln_b, xn_h, vecn, vecn_h, N);
  k_node_qkv<<<ntiles, 128, 0, stream>>>(xn_h, vecn_h, pk, bq, bk, bv,
                                         qkv, vdot, vec3, N);
  size_t zn = (size_t)N * 4 * H;
  int zblocks = (int)((zn + 255) / 256); if (zblocks > 65535) zblocks = 65535;
  k_zero<<<zblocks, 256, 0, stream>>>(x_agg, zn);
  k_edge<<<etiles, 128, 0, stream>>>(qkv, vecn, vecn_h, f_ij, eidx, r_ij, d_ij,
                                     pk, bdk, bdv, bf, bs,
                                     x_agg, vec_agg, dfij, E);
  k_node_out<<<ntiles, 128, 0, stream>>>(x_agg, vec_agg, vdot, vec3, pk, bo, dx, dvec, N);
}
